// IRnet_53017076302299
// MI455X (gfx1250) — matrix operand order repaired and checked by reading, not compiled, not run on hardware
//
#include <hip/hip_runtime.h>

// ---------------------------------------------------------------------------
// Problem constants (from reference setup_inputs)
// ---------------------------------------------------------------------------
#define BB   8
#define HH   512
#define WW   512
#define HP   514        // padded
#define WP   514
#define PP   32         // control grid P
#define NPTS (BB * PP * PP)   // 8192 control points
#define HD   64

typedef __attribute__((ext_vector_type(16))) _Float16 v16h;
typedef __attribute__((ext_vector_type(8)))  float    v8f;

union V16H { uint4 u[2]; v16h h; };
union HBits { _Float16 h; unsigned short s; };

// ---------------------------------------------------------------------------
// Zero fill (for padded activation buffers)
// ---------------------------------------------------------------------------
__global__ __launch_bounds__(256) void reg_fill_zero4(uint4* __restrict__ p, long long n4) {
    long long i = (long long)blockIdx.x * 256 + threadIdx.x;
    if (i < n4) { uint4 z; z.x = z.y = z.z = z.w = 0u; p[i] = z; }
}

// ---------------------------------------------------------------------------
// Pack conv weights (Cout=64, Cin, 3,3) f32 -> WMMA B-matrix layout fp16.
// Layout: u32 index = (((tap*KC + kc)*4 + nt)*32 + lane)*8 + vgpr
//   N    = nt*16 + (lane & 15)
//   koff = kc*32 + ((lane & 16) ? 16 : 0)
//   Kpair = koff + 2*vgpr (+1)   -> c_in
// Each u32 packs (K even -> [15:0], K odd -> [31:16]).
// ---------------------------------------------------------------------------
__global__ __launch_bounds__(256) void reg_pack_w(const float* __restrict__ w,
                                                  unsigned int* __restrict__ wpack,
                                                  int Cin, int KC) {
    int idx = blockIdx.x * 256 + threadIdx.x;
    int total = 9 * KC * 4 * 32 * 8;
    if (idx >= total) return;
    int vg   = idx & 7;
    int lane = (idx >> 3) & 31;
    int nt   = (idx >> 8) & 3;
    int rest = idx >> 10;            // tap*KC + kc
    int kc   = rest % KC;
    int tap  = rest / KC;
    int ky = tap / 3, kx = tap % 3;
    int N    = nt * 16 + (lane & 15);
    int koff = kc * 32 + ((lane & 16) ? 16 : 0);
    int c0 = koff + vg * 2;
    int c1 = c0 + 1;
    float w0 = w[((N * Cin + c0) * 3 + ky) * 3 + kx];
    float w1 = w[((N * Cin + c1) * 3 + ky) * 3 + kx];
    HBits a, b; a.h = (_Float16)w0; b.h = (_Float16)w1;
    wpack[idx] = (unsigned int)a.s | ((unsigned int)b.s << 16);
}

// ---------------------------------------------------------------------------
// Conv1: (B,2,H,W) -> relu -> x1p fp16 padded (B,514,514,32) channels-last.
// K=18 is tiny: plain VALU kernel, weights staged in LDS.
// ---------------------------------------------------------------------------
__global__ __launch_bounds__(256) void reg_conv1(const float* __restrict__ mov,
                                                 const float* __restrict__ fix,
                                                 const float* __restrict__ w,
                                                 const float* __restrict__ bias,
                                                 _Float16* __restrict__ x1p) {
    __shared__ float wl[576];
    __shared__ float bl[32];
    int t = threadIdx.x;
    for (int i = t; i < 576; i += 256) wl[i] = w[i];
    if (t < 32) bl[t] = bias[t];
    __syncthreads();

    int idx = blockIdx.x * 256 + t;           // B*H*W threads
    int b   = idx >> 18;
    int rem = idx & 262143;
    int y   = rem >> 9;
    int x   = rem & 511;

    float in0[9], in1[9];
    for (int ky = 0; ky < 3; ++ky) {
        for (int kx = 0; kx < 3; ++kx) {
            int yy = y + ky - 1, xx = x + kx - 1;
            bool ok = (yy >= 0) && (yy < HH) && (xx >= 0) && (xx < WW);
            size_t off = ((size_t)b * HH + (ok ? yy : 0)) * WW + (ok ? xx : 0);
            in0[ky * 3 + kx] = ok ? mov[off] : 0.f;
            in1[ky * 3 + kx] = ok ? fix[off] : 0.f;
        }
    }
    float acc[32];
    for (int c = 0; c < 32; ++c) acc[c] = bl[c];
    for (int k = 0; k < 9; ++k) {
        float a0 = in0[k], a1 = in1[k];
        for (int c = 0; c < 32; ++c)
            acc[c] += wl[(c * 2 + 0) * 9 + k] * a0 + wl[(c * 2 + 1) * 9 + k] * a1;
    }
    union { _Float16 h[32]; uint4 u[4]; } o;
    for (int c = 0; c < 32; ++c) {
        float v = acc[c];
        o.h[c] = (_Float16)(v > 0.f ? v : 0.f);
    }
    _Float16* dst = x1p + (((size_t)b * HP + (y + 1)) * WP + (x + 1)) * 32;
    uint4* d4 = (uint4*)dst;
    d4[0] = o.u[0]; d4[1] = o.u[1]; d4[2] = o.u[2]; d4[3] = o.u[3];
}

// ---------------------------------------------------------------------------
// Conv2: x1p (B,514,514,32) f16 -> relu -> x2p (B,514,514,64) f16.
// Implicit GEMM: per wave 16 pixels x 64 outch, K = 9 taps * 32.
// 9 taps * 4 N-tiles = 36 v_wmma_f32_16x16x32_f16 per wave.
// ---------------------------------------------------------------------------
__global__ __launch_bounds__(256) void reg_conv2_wmma(const _Float16* __restrict__ x1p,
                                                      const unsigned int* __restrict__ wpack,
                                                      const float* __restrict__ bias,
                                                      _Float16* __restrict__ x2p) {
    __shared__ uint4 wlds[2304];                       // 36 KB packed weights
    __shared__ __align__(16) _Float16 ostage[128 * 64]; // 16 KB output transpose
    const int t = threadIdx.x;
    const uint4* wsrc = (const uint4*)wpack;
    for (int i = t; i < 2304; i += 256) wlds[i] = wsrc[i];
    __syncthreads();

    const int lane = t & 31, wave = t >> 5;
    const int xblock = blockIdx.x * 128;               // 4 blocks per row
    const int y = blockIdx.y;
    const int b = blockIdx.z;
    const int xw = xblock + wave * 16;
    const int m = lane & 15;
    const int kh = (lane >> 4) * 16;                   // half-offset for K-half select

    v8f acc0 = {}, acc1 = {}, acc2 = {}, acc3 = {};
    for (int tap = 0; tap < 9; ++tap) {
        const int ky = tap / 3, kx = tap % 3;
        const _Float16* ap = x1p + (((size_t)b * HP + (y + ky)) * WP + (xw + kx + m)) * 32 + kh;
        V16H a;
        a.u[0] = *(const uint4*)(ap);
        a.u[1] = *(const uint4*)(ap + 8);
        const int bi = (tap * 4) * 64 + lane * 2;
        V16H b0, b1, b2, b3;
        b0.u[0] = wlds[bi];           b0.u[1] = wlds[bi + 1];
        b1.u[0] = wlds[bi + 64];      b1.u[1] = wlds[bi + 65];
        b2.u[0] = wlds[bi + 128];     b2.u[1] = wlds[bi + 129];
        b3.u[0] = wlds[bi + 192];     b3.u[1] = wlds[bi + 193];
        acc0 = __builtin_amdgcn_wmma_f32_16x16x32_f16(false, a.h, false, b0.h, (short)0, acc0, false, false);
        acc1 = __builtin_amdgcn_wmma_f32_16x16x32_f16(false, a.h, false, b1.h, (short)0, acc1, false, false);
        acc2 = __builtin_amdgcn_wmma_f32_16x16x32_f16(false, a.h, false, b2.h, (short)0, acc2, false, false);
        acc3 = __builtin_amdgcn_wmma_f32_16x16x32_f16(false, a.h, false, b3.h, (short)0, acc3, false, false);
    }
    // bias + relu + transpose stage  (C layout: vgpr r -> M=r (+8 for hi lanes), N=lane&15)
    const int mhi = (lane >> 4) << 3;
    v8f accs[4] = {acc0, acc1, acc2, acc3};
    for (int nt = 0; nt < 4; ++nt) {
        const int c = nt * 16 + m;
        const float bc = bias[c];
        for (int r = 0; r < 8; ++r) {
            float v = accs[nt][r] + bc;
            v = v > 0.f ? v : 0.f;
            ostage[(wave * 16 + r + mhi) * 64 + c] = (_Float16)v;
        }
    }
    __syncthreads();
    // coalesced writeback: thread -> (pixel, 64B half)
    const int p = t >> 1, hhalf = (t & 1) * 32;
    const uint4* s4 = (const uint4*)(&ostage[p * 64 + hhalf]);
    _Float16* dst = x2p + (((size_t)b * HP + (y + 1)) * WP + (xblock + p + 1)) * 64 + hhalf;
    uint4* d4 = (uint4*)dst;
    d4[0] = s4[0]; d4[1] = s4[1]; d4[2] = s4[2]; d4[3] = s4[3];
}

// ---------------------------------------------------------------------------
// Conv3: x2p (B,514,514,64) f16 -> feat (B,512,512,64) f16 (no relu).
// 9 taps * 2 K-chunks * 4 N-tiles = 72 WMMAs per wave.
// ---------------------------------------------------------------------------
__global__ __launch_bounds__(256) void reg_conv3_wmma(const _Float16* __restrict__ x2p,
                                                      const unsigned int* __restrict__ wpack,
                                                      const float* __restrict__ bias,
                                                      _Float16* __restrict__ feat) {
    __shared__ uint4 wlds[4608];                        // 72 KB packed weights
    __shared__ __align__(16) _Float16 ostage[128 * 64]; // 16 KB
    const int t = threadIdx.x;
    const uint4* wsrc = (const uint4*)wpack;
    for (int i = t; i < 4608; i += 256) wlds[i] = wsrc[i];
    __syncthreads();

    const int lane = t & 31, wave = t >> 5;
    const int xblock = blockIdx.x * 128;
    const int y = blockIdx.y;
    const int b = blockIdx.z;
    const int xw = xblock + wave * 16;
    const int m = lane & 15;
    const int kh = (lane >> 4) * 16;

    v8f acc0 = {}, acc1 = {}, acc2 = {}, acc3 = {};
    for (int tap = 0; tap < 9; ++tap) {
        const int ky = tap / 3, kx = tap % 3;
        const _Float16* prow = x2p + (((size_t)b * HP + (y + ky)) * WP + (xw + kx + m)) * 64;
        for (int kc = 0; kc < 2; ++kc) {
            const _Float16* ap = prow + kc * 32 + kh;
            V16H a;
            a.u[0] = *(const uint4*)(ap);
            a.u[1] = *(const uint4*)(ap + 8);
            const int bi = ((tap * 2 + kc) * 4) * 64 + lane * 2;
            V16H b0, b1, b2, b3;
            b0.u[0] = wlds[bi];       b0.u[1] = wlds[bi + 1];
            b1.u[0] = wlds[bi + 64];  b1.u[1] = wlds[bi + 65];
            b2.u[0] = wlds[bi + 128]; b2.u[1] = wlds[bi + 129];
            b3.u[0] = wlds[bi + 192]; b3.u[1] = wlds[bi + 193];
            acc0 = __builtin_amdgcn_wmma_f32_16x16x32_f16(false, a.h, false, b0.h, (short)0, acc0, false, false);
            acc1 = __builtin_amdgcn_wmma_f32_16x16x32_f16(false, a.h, false, b1.h, (short)0, acc1, false, false);
            acc2 = __builtin_amdgcn_wmma_f32_16x16x32_f16(false, a.h, false, b2.h, (short)0, acc2, false, false);
            acc3 = __builtin_amdgcn_wmma_f32_16x16x32_f16(false, a.h, false, b3.h, (short)0, acc3, false, false);
        }
    }
    const int mhi = (lane >> 4) << 3;
    v8f accs[4] = {acc0, acc1, acc2, acc3};
    for (int nt = 0; nt < 4; ++nt) {
        const int c = nt * 16 + m;
        const float bc = bias[c];
        for (int r = 0; r < 8; ++r) {
            float v = accs[nt][r] + bc;   // no relu on conv3
            ostage[(wave * 16 + r + mhi) * 64 + c] = (_Float16)v;
        }
    }
    __syncthreads();
    const int p = t >> 1, hhalf = (t & 1) * 32;
    const uint4* s4 = (const uint4*)(&ostage[p * 64 + hhalf]);
    _Float16* dst = feat + (((size_t)b * HH + y) * WW + (xblock + p)) * 64 + hhalf;
    uint4* d4 = (uint4*)dst;
    d4[0] = s4[0]; d4[1] = s4[1]; d4[2] = s4[2]; d4[3] = s4[3];
}

// ---------------------------------------------------------------------------
// Normalized control points
// ---------------------------------------------------------------------------
__global__ __launch_bounds__(256) void reg_normcp(const int* __restrict__ cps,
                                                  float* __restrict__ normcp) {
    int p = blockIdx.x * 256 + threadIdx.x;
    if (p >= NPTS) return;
    normcp[p * 2 + 0] = (float)cps[p * 2 + 0] / (float)HH * 2.f - 1.f;
    normcp[p * 2 + 1] = (float)cps[p * 2 + 1] / (float)WW * 2.f - 1.f;
}

// ---------------------------------------------------------------------------
// Bicubic grid sample (A = -0.75, reflect padding). One wave per point,
// lane handles channels 2*lane, 2*lane+1. Writes into dcat[:, 64:128].
// ---------------------------------------------------------------------------
__device__ __forceinline__ float reg_reflect(float x) {
    const float span = 511.f;
    x = fabsf(x);
    x = fmodf(x, 2.f * span);
    x = (x > span) ? (2.f * span - x) : x;
    return fminf(fmaxf(x, 0.f), span);
}
__device__ __forceinline__ float reg_cc1(float x) { return (1.25f * x - 2.25f) * x * x + 1.f; }
__device__ __forceinline__ float reg_cc2(float x) { return ((-0.75f * x + 3.75f) * x - 6.f) * x + 3.f; }

__global__ __launch_bounds__(256) void reg_sample_bicubic(const _Float16* __restrict__ feat,
                                                          const float* __restrict__ normcp,
                                                          float* __restrict__ dcat) {
    const int t = threadIdx.x, lane = t & 31, wave = t >> 5;
    const int p = blockIdx.x * 8 + wave;
    const int b = p >> 10;
    const float gx = normcp[p * 2 + 0], gy = normcp[p * 2 + 1];
    const float ix = reg_reflect((gx + 1.f) * 0.5f * (float)(WW - 1));
    const float iy = reg_reflect((gy + 1.f) * 0.5f * (float)(HH - 1));
    const float fx = floorf(ix), fy = floorf(iy);
    const float tx = ix - fx, ty = iy - fy;
    int xs[4], ys[4];
    float wx[4], wy[4];
    for (int j = 0; j < 4; ++j) {
        xs[j] = (int)reg_reflect(fx + (float)(j - 1));
        ys[j] = (int)reg_reflect(fy + (float)(j - 1));
    }
    wx[0] = reg_cc2(tx + 1.f); wx[1] = reg_cc1(tx); wx[2] = reg_cc1(1.f - tx); wx[3] = reg_cc2(2.f - tx);
    wy[0] = reg_cc2(ty + 1.f); wy[1] = reg_cc1(ty); wy[2] = reg_cc1(1.f - ty); wy[3] = reg_cc2(2.f - ty);

    const int c = lane * 2;
    float a0 = 0.f, a1 = 0.f;
    for (int i = 0; i < 4; ++i) {
        for (int j = 0; j < 4; ++j) {
            const _Float16* fp = feat + (((size_t)b * HH + ys[i]) * WW + xs[j]) * 64 + c;
            const float wgt = wy[i] * wx[j];
            a0 += wgt * (float)fp[0];
            a1 += wgt * (float)fp[1];
        }
    }
    dcat[(size_t)p * 128 + 64 + c]     = a0;
    dcat[(size_t)p * 128 + 64 + c + 1] = a1;
}

// ---------------------------------------------------------------------------
// Generic dense layer: one point per block, thread j computes out[p][j].
// ---------------------------------------------------------------------------
__global__ __launch_bounds__(128) void reg_dense(const float* __restrict__ in, int in_stride, int in_dim,
                                                 const float* __restrict__ w, const float* __restrict__ bias,
                                                 float* __restrict__ out, int out_stride, int out_dim,
                                                 int do_relu) {
    __shared__ float xin[128];
    const int p = blockIdx.x, t = threadIdx.x;
    if (t < in_dim) xin[t] = in[(size_t)p * in_stride + t];
    __syncthreads();
    if (t < out_dim) {
        float acc = bias[t];
        const float* wr = w + (size_t)t * in_dim;
        for (int i = 0; i < in_dim; ++i) acc += xin[i] * wr[i];
        if (do_relu) acc = fmaxf(acc, 0.f);
        out[(size_t)p * out_stride + t] = acc;
    }
}

// ---------------------------------------------------------------------------
// Separable cubic B-spline upsample of disp (B,32,32,2) -> out (B,512,512,2)
// ---------------------------------------------------------------------------
__global__ __launch_bounds__(256) void reg_bspline(const float* __restrict__ disp,
                                                   float* __restrict__ out) {
    int idx = blockIdx.x * 256 + threadIdx.x;   // B*H*W
    int b   = idx >> 18;
    int rem = idx & 262143;
    int h   = rem >> 9;
    int w   = rem & 511;

    float wh[4], wwt[4];
    int ih[4], iw[4];
    {
        float u = (float)h * (1.f / 16.f);
        float i0 = floorf(u);
        float tt = u - i0;
        float t2 = tt * tt, t3 = t2 * tt, om = 1.f - tt;
        wh[0] = om * om * om * (1.f / 6.f);
        wh[1] = (3.f * t3 - 6.f * t2 + 4.f) * (1.f / 6.f);
        wh[2] = (-3.f * t3 + 3.f * t2 + 3.f * tt + 1.f) * (1.f / 6.f);
        wh[3] = t3 * (1.f / 6.f);
        for (int k = 0; k < 4; ++k) ih[k] = min(max((int)i0 + k - 1, 0), PP - 1);
    }
    {
        float u = (float)w * (1.f / 16.f);
        float i0 = floorf(u);
        float tt = u - i0;
        float t2 = tt * tt, t3 = t2 * tt, om = 1.f - tt;
        wwt[0] = om * om * om * (1.f / 6.f);
        wwt[1] = (3.f * t3 - 6.f * t2 + 4.f) * (1.f / 6.f);
        wwt[2] = (-3.f * t3 + 3.f * t2 + 3.f * tt + 1.f) * (1.f / 6.f);
        wwt[3] = t3 * (1.f / 6.f);
        for (int k = 0; k < 4; ++k) iw[k] = min(max((int)i0 + k - 1, 0), PP - 1);
    }
    float a0 = 0.f, a1 = 0.f;
    for (int k = 0; k < 4; ++k) {
        for (int l = 0; l < 4; ++l) {
            const float wgt = wh[k] * wwt[l];
            const float* d = disp + (((size_t)b * PP + ih[k]) * PP + iw[l]) * 2;
            a0 += wgt * d[0];
            a1 += wgt * d[1];
        }
    }
    float* o = out + (((size_t)b * HH + h) * WW + w) * 2;
    o[0] = a0; o[1] = a1;
}

// ---------------------------------------------------------------------------
// Host launcher
// ---------------------------------------------------------------------------
extern "C" void kernel_launch(void* const* d_in, const int* in_sizes, int n_in,
                              void* d_out, int out_size, void* d_ws, size_t ws_size,
                              hipStream_t stream) {
    (void)in_sizes; (void)n_in; (void)out_size; (void)ws_size;
    const float* mov   = (const float*)d_in[0];
    const float* fix   = (const float*)d_in[1];
    const int*   cps   = (const int*)  d_in[2];
    const float* ce_w1 = (const float*)d_in[3];  const float* ce_b1 = (const float*)d_in[4];
    const float* ce_w2 = (const float*)d_in[5];  const float* ce_b2 = (const float*)d_in[6];
    const float* ce_w3 = (const float*)d_in[7];  const float* ce_b3 = (const float*)d_in[8];
    const float* cv_w1 = (const float*)d_in[9];  const float* cv_b1 = (const float*)d_in[10];
    const float* cv_w2 = (const float*)d_in[11]; const float* cv_b2 = (const float*)d_in[12];
    const float* cv_w3 = (const float*)d_in[13]; const float* cv_b3 = (const float*)d_in[14];
    const float* d_w1  = (const float*)d_in[15]; const float* d_b1  = (const float*)d_in[16];
    const float* d_w2  = (const float*)d_in[17]; const float* d_b2  = (const float*)d_in[18];
    const float* d_w3  = (const float*)d_in[19]; const float* d_b3  = (const float*)d_in[20];
    float* out = (float*)d_out;

    // Workspace carve-up (256B aligned)
    char* ws = (char*)d_ws;
    size_t off = 0;
    auto alloc = [&](size_t bytes) -> void* {
        void* p = ws + off;
        off = (off + bytes + 255) & ~(size_t)255;
        return p;
    };
    const size_t x1p_bytes  = (size_t)BB * HP * WP * 32 * 2;   // ~135 MB
    const size_t x2p_bytes  = (size_t)BB * HP * WP * 64 * 2;   // ~271 MB
    const size_t feat_bytes = (size_t)BB * HH * WW * 64 * 2;   // ~268 MB
    _Float16* x1p  = (_Float16*)alloc(x1p_bytes);
    _Float16* x2p  = (_Float16*)alloc(x2p_bytes);
    _Float16* feat = (_Float16*)alloc(feat_bytes);
    unsigned int* wpack2 = (unsigned int*)alloc(9 * 1 * 4 * 32 * 8 * 4);   // 36 KB
    unsigned int* wpack3 = (unsigned int*)alloc(9 * 2 * 4 * 32 * 8 * 4);   // 72 KB
    float* normcp = (float*)alloc((size_t)NPTS * 2 * 4);
    float* h1     = (float*)alloc((size_t)NPTS * HD * 4);
    float* h2     = (float*)alloc((size_t)NPTS * HD * 4);
    float* dcat   = (float*)alloc((size_t)NPTS * 128 * 4);
    float* dh1    = (float*)alloc((size_t)NPTS * 128 * 4);
    float* dh2    = (float*)alloc((size_t)NPTS * 128 * 4);
    float* disp   = (float*)alloc((size_t)NPTS * 2 * 4);

    // 1) zero the padded activation buffers (borders = SAME zero-padding)
    {
        long long n4 = (long long)(x1p_bytes / 16);
        reg_fill_zero4<<<(unsigned)((n4 + 255) / 256), 256, 0, stream>>>((uint4*)x1p, n4);
        n4 = (long long)(x2p_bytes / 16);
        reg_fill_zero4<<<(unsigned)((n4 + 255) / 256), 256, 0, stream>>>((uint4*)x2p, n4);
    }
    // 2) pack conv2/conv3 weights into WMMA B layout
    reg_pack_w<<<(9 * 1 * 4 * 32 * 8 + 255) / 256, 256, 0, stream>>>(cv_w2, wpack2, 32, 1);
    reg_pack_w<<<(9 * 2 * 4 * 32 * 8 + 255) / 256, 256, 0, stream>>>(cv_w3, wpack3, 64, 2);
    // 3) conv1 (VALU) -> x1p
    reg_conv1<<<(BB * HH * WW) / 256, 256, 0, stream>>>(mov, fix, cv_w1, cv_b1, x1p);
    // 4) conv2 / conv3 (WMMA implicit GEMM)
    dim3 cgrid(WW / 128, HH, BB);
    reg_conv2_wmma<<<cgrid, 256, 0, stream>>>(x1p, wpack2, cv_b2, x2p);
    reg_conv3_wmma<<<cgrid, 256, 0, stream>>>(x2p, wpack3, cv_b3, feat);
    // 5) coordinate embedding MLP -> dcat[:, 0:64]
    reg_normcp<<<(NPTS + 255) / 256, 256, 0, stream>>>(cps, normcp);
    reg_dense<<<NPTS, 128, 0, stream>>>(normcp, 2, 2,   ce_w1, ce_b1, h1,   HD,  HD, 1);
    reg_dense<<<NPTS, 128, 0, stream>>>(h1,     HD, HD, ce_w2, ce_b2, h2,   HD,  HD, 1);
    reg_dense<<<NPTS, 128, 0, stream>>>(h2,     HD, HD, ce_w3, ce_b3, dcat, 128, HD, 0);
    // 6) bicubic sampling of feat -> dcat[:, 64:128]
    reg_sample_bicubic<<<NPTS / 8, 256, 0, stream>>>(feat, normcp, dcat);
    // 7) displacement MLP -> disp
    reg_dense<<<NPTS, 128, 0, stream>>>(dcat, 128, 128, d_w1, d_b1, dh1,  128, 128, 1);
    reg_dense<<<NPTS, 128, 0, stream>>>(dh1,  128, 128, d_w2, d_b2, dh2,  128, 128, 1);
    reg_dense<<<NPTS, 128, 0, stream>>>(dh2,  128, 128, d_w3, d_b3, disp, 2,   2,   0);
    // 8) B-spline dense field -> output (B,512,512,2) f32
    reg_bspline<<<(BB * HH * WW) / 256, 256, 0, stream>>>(disp, out);
}
